// GraphScaledDotProductAttention_6262062318237
// MI455X (gfx1250) — matrix operand order repaired and checked by reading, not compiled, not run on hardware
//
#include <hip/hip_runtime.h>
#include <hip/hip_bf16.h>

// ---------------------------------------------------------------------------
// Fused graph attention for MI455X (gfx1250, wave32, WMMA), v3.
//   scores  = (Q/sqrt(64)) @ K^T + bias - 0.5*g^2    (gauss denom (0.5*2)^2 = 1)
//   weights = softmax(scores)  -> d_out[1] (fp32, 536 MB)
//   out     = weights @ V      -> d_out[0] (fp32)
//
// Design (roofline: ~1.68 GB mandatory HBM traffic -> ~72 us @ 23.3 TB/s;
// bf16 XDL math ~34 GFLOP total is negligible, fp32 16x16x4 WMMA would not be):
//  * 256 threads = 8 waves per block on one 16-row q-tile; ~129 KB LDS
//    -> 2 blocks/WGP -> 16 waves/WGP to drive HBM.
//  * Phase 1: key-tiles striped across waves; bf16 WMMA 16x16x32, f32 acc.
//  * Phase 2: softmax with ternary max (v_max_num, no NaN gunk) and float4
//    LDS scans (ds_load_b128). Unnormalized exp kept in LDS.
//  * Phase 3: float4 normalized weight write-out (512B/wave stores).
//  * Phase 4: each wave owns a contiguous 256-key chunk; f32->bf16 on the fly
//    A-fragments (float4 LDS reads, padded stride kills bank conflicts);
//    V B-fragments straight from global (2x64B segments per element, L2-hot);
//    cross-wave partial reduction reuses the score buffer.
// ---------------------------------------------------------------------------

#define S_LEN 2048
#define DK    64
#define TQ    16
#define NKT   (S_LEN / 16)
#define BH    32              // B*H = 4*8
#define NW    8               // waves per block
#define SROW  2064            // padded f32 row stride (2064 % 64 = 16)

typedef __attribute__((ext_vector_type(16))) __bf16 v16bf;
typedef __attribute__((ext_vector_type(8)))  float  v8f;

__device__ __forceinline__ float vmaxf(float a, float b) { return a > b ? a : b; }

__device__ __forceinline__ void load_a_frag_f32(v16bf& a, const float* __restrict__ rowp,
                                                int abase, float scl) {
#pragma unroll
  for (int e = 0; e < 8; ++e) {
    a[e]     = (__bf16)(rowp[abase + e] * scl);
    a[e + 8] = (__bf16)(rowp[abase + 16 + e] * scl);
  }
}

__global__ __launch_bounds__(NW * 32)
void fused_graph_attn(const float* __restrict__ q, const float* __restrict__ kmat,
                      const float* __restrict__ v, const float* __restrict__ bias,
                      const float* __restrict__ gbias,
                      float* __restrict__ out_o, float* __restrict__ out_w) {
  extern __shared__ char smem[];
  float* slds = (float*)smem;                        // [16][SROW] f32  (~129 KB)
  float* invl = (float*)(smem + TQ * SROW * 4);      // [16] 1/rowsum

  const int tid   = threadIdx.x;
  const int wv    = tid >> 5;             // wave id 0..7
  const int lane  = tid & 31;
  const int ln15  = lane & 15;            // A-row M / B,C,D column N
  const int mbase = (lane >> 4) * 8;      // C/D rows: mbase..mbase+7
  const int abase = (lane < 16) ? 0 : 8;  // A-fragment K base (16-bit A layout)
  const int bbase = (lane < 16) ? 0 : 16; // B-fragment K base

  const int bh = blockIdx.y;              // b*H + h
  const int q0 = blockIdx.x * TQ;

  const size_t qkvBase = (size_t)bh * S_LEN * DK;
  const size_t bBase   = (size_t)bh * S_LEN * S_LEN;

  // ---- Q fragments (per-wave copy; 4 KB, L2-hot) ----
  v16bf aq0, aq1;
  {
    const float* qrow = q + qkvBase + (size_t)(q0 + ln15) * DK;
    load_a_frag_f32(aq0, qrow,      abase, 0.125f);   // 1/sqrt(64)
    load_a_frag_f32(aq1, qrow + 32, abase, 0.125f);
  }

  // ---- Phase 1: scores; key-tiles striped across the 8 waves ----
  for (int kt = wv; kt < NKT; kt += NW) {
    v16bf bk0, bk1;  // B = K^T: element (kd, n) = K[kt*16+n][c*32+kd]
    const float* krow = kmat + qkvBase + (size_t)(kt * 16 + ln15) * DK;
#pragma unroll
    for (int e = 0; e < 16; ++e) {
      bk0[e] = (__bf16)krow[abase + (e < 8 ? e : e + 8)];
      bk1[e] = (__bf16)krow[32 + abase + (e < 8 ? e : e + 8)];
    }
    v8f acc = {0.f, 0.f, 0.f, 0.f, 0.f, 0.f, 0.f, 0.f};
    acc = __builtin_amdgcn_wmma_f32_16x16x32_bf16(false, aq0, false, bk0, (short)0, acc, false, false);
    acc = __builtin_amdgcn_wmma_f32_16x16x32_bf16(false, aq1, false, bk1, (short)0, acc, false, false);

    const int col = kt * 16 + ln15;
    {  // unconditional prefetch of a (uniformly clamped) future tile: no exec games
      const int ktn = (kt + NW < NKT) ? (kt + NW) : kt;
      const size_t pi = bBase + (size_t)(q0 + mbase) * S_LEN + ktn * 16 + ln15;
      __builtin_prefetch(bias + pi, 0, 0);    // global_prefetch_b8 (GL2)
      __builtin_prefetch(gbias + pi, 0, 0);
    }
#pragma unroll
    for (int r = 0; r < 8; ++r) {
      const int m = mbase + r;
      const size_t bi = bBase + (size_t)(q0 + m) * S_LEN + col;
      const float g = gbias[bi];
      slds[m * SROW + col] = acc[r] + bias[bi] - 0.5f * g * g;
    }
  }
  __syncthreads();

  // ---- Phase 2: row softmax; 16 threads/row, float4 scans over 128 cols ----
  {
    const int rr = tid >> 4;                       // row 0..15 (row's threads share a wave)
    float4* s4 = (float4*)(slds + rr * SROW + (tid & 15) * 128);
    float mval = -3.0e38f;
#pragma unroll 4
    for (int i = 0; i < 32; ++i) {
      const float4 x = s4[i];                      // ds_load_b128
      mval = vmaxf(mval, vmaxf(vmaxf(x.x, x.y), vmaxf(x.z, x.w)));
    }
#pragma unroll
    for (int off = 1; off < 16; off <<= 1) mval = vmaxf(mval, __shfl_xor(mval, off, 32));
    float ssum = 0.f;
#pragma unroll 4
    for (int i = 0; i < 32; ++i) {
      float4 x = s4[i];
      x.x = __expf(x.x - mval); x.y = __expf(x.y - mval);
      x.z = __expf(x.z - mval); x.w = __expf(x.w - mval);
      s4[i] = x;                                   // keep UNNORMALIZED exp in LDS
      ssum += (x.x + x.y) + (x.z + x.w);
    }
#pragma unroll
    for (int off = 1; off < 16; off <<= 1) ssum += __shfl_xor(ssum, off, 32);
    if ((tid & 15) == 0) invl[rr] = 1.0f / ssum;
  }
  __syncthreads();

  // ---- Phase 3: normalized fp32 weights, float4 coalesced (4 KB per block-step) ----
  for (int r = 0; r < TQ; ++r) {
    const float il = invl[r];
    const float4* sr4 = (const float4*)(slds + r * SROW);
    float4* gw4 = (float4*)(out_w + bBase + (size_t)(q0 + r) * S_LEN);
#pragma unroll
    for (int c = tid; c < S_LEN / 4; c += NW * 32) {
      float4 x = sr4[c];
      x.x *= il; x.y *= il; x.z *= il; x.w *= il;
      gw4[c] = x;
    }
  }

  // ---- Phase 4: out = exp @ V; each wave owns a contiguous 256-key chunk ----
  v8f accd[4];
#pragma unroll
  for (int dt = 0; dt < 4; ++dt) accd[dt] = (v8f){0.f, 0.f, 0.f, 0.f, 0.f, 0.f, 0.f, 0.f};

  const int kk0 = wv * (S_LEN / NW);
  for (int kk = kk0; kk < kk0 + S_LEN / NW; kk += 32) {
    // A = exp-weights 16x32 chunk: two float4 pairs (16B-aligned; padded stride
    // -> 16 lanes land in distinct banks), f32 -> bf16 on the fly.
    v16bf aw;
    {
      const float4* sa = (const float4*)(slds + ln15 * SROW + kk + abase);
      const float4 x0 = sa[0], x1 = sa[1];         // K = abase .. abase+7
      const float4* sb = (const float4*)(slds + ln15 * SROW + kk + abase + 16);
      const float4 y0 = sb[0], y1 = sb[1];         // K = abase+16 .. abase+23
      aw[0]  = (__bf16)x0.x; aw[1]  = (__bf16)x0.y; aw[2]  = (__bf16)x0.z; aw[3]  = (__bf16)x0.w;
      aw[4]  = (__bf16)x1.x; aw[5]  = (__bf16)x1.y; aw[6]  = (__bf16)x1.z; aw[7]  = (__bf16)x1.w;
      aw[8]  = (__bf16)y0.x; aw[9]  = (__bf16)y0.y; aw[10] = (__bf16)y0.z; aw[11] = (__bf16)y0.w;
      aw[12] = (__bf16)y1.x; aw[13] = (__bf16)y1.y; aw[14] = (__bf16)y1.z; aw[15] = (__bf16)y1.w;
    }
#pragma unroll
    for (int dt = 0; dt < 4; ++dt) {
      v16bf bv;  // B element (kd, n) = V[kk+kd][dt*16+n]; per-e: two 64B segments
      const float* vp = v + qkvBase + (size_t)(kk + abase) * DK + dt * 16 + ln15;
#pragma unroll
      for (int e = 0; e < 16; ++e) bv[e] = (__bf16)vp[(size_t)(e < 8 ? e : e + 8) * DK];
      accd[dt] = __builtin_amdgcn_wmma_f32_16x16x32_bf16(false, aw, false, bv, (short)0, accd[dt], false, false);
    }
  }
  __syncthreads();   // all waves done reading slds -> safe to reuse as partial buffer

  // ---- Cross-wave reduction of 16x64 partials (reuses slds region) ----
  float* part = slds;  // [NW][16][64]
#pragma unroll
  for (int dt = 0; dt < 4; ++dt) {
#pragma unroll
    for (int r = 0; r < 8; ++r) {
      part[(wv * TQ + mbase + r) * DK + dt * 16 + ln15] = accd[dt][r];
    }
  }
  __syncthreads();

#pragma unroll
  for (int j = 0; j < (TQ * DK) / (NW * 32); ++j) {   // 4 elements per thread
    const int idx = j * (NW * 32) + tid;
    const int m = idx >> 6, d = idx & 63;
    float s = 0.f;
#pragma unroll
    for (int ww = 0; ww < NW; ++ww) s += part[(ww * TQ + m) * DK + d];
    out_o[qkvBase + (size_t)(q0 + m) * DK + d] = s * invl[m];
  }
}

extern "C" void kernel_launch(void* const* d_in, const int* in_sizes, int n_in,
                              void* d_out, int out_size, void* d_ws, size_t ws_size,
                              hipStream_t stream) {
  (void)in_sizes; (void)n_in; (void)d_ws; (void)ws_size; (void)out_size;
  const float* q     = (const float*)d_in[0];
  const float* k     = (const float*)d_in[1];
  const float* v     = (const float*)d_in[2];
  const float* bias  = (const float*)d_in[3];
  const float* gbias = (const float*)d_in[4];

  float* out_o = (float*)d_out;                      // [4,8,2048,64]
  float* out_w = out_o + (size_t)BH * S_LEN * DK;    // [4,8,2048,2048]

  const size_t lds = (size_t)TQ * SROW * 4 + 16 * sizeof(float);  // ~129 KB
  (void)hipFuncSetAttribute((const void*)fused_graph_attn,
                            hipFuncAttributeMaxDynamicSharedMemorySize, (int)lds);

  dim3 grid(S_LEN / TQ, BH);   // 128 q-tiles x 32 (b,h)
  fused_graph_attn<<<grid, NW * 32, lds, stream>>>(q, k, v, bias, gbias, out_o, out_w);
}
